// BitwiseBasicBlock_18648747999435
// MI455X (gfx1250) — 8-bit matrix operand order repaired and checked by reading, compiled, not run on hardware
//
#include <hip/hip_runtime.h>
#include <stdint.h>
#include <stddef.h>

typedef __attribute__((ext_vector_type(4))) int          v4i;
typedef __attribute__((ext_vector_type(8))) int          v8i;
typedef __attribute__((ext_vector_type(4))) unsigned int v4u;
typedef __attribute__((ext_vector_type(4))) float        v4f;

#define NN 32
#define CC_ 256
#define HH_ 56
#define WW_ 56
#define HP 58
#define WP 58
#define NPIX (NN*HH_*WW_)
#define KSTEPS 36
#define LDSW_BYTES (2*36*1024)
#define LDSX_ROW (66*256)
#define LDSX_BYTES (3*LDSX_ROW)

__device__ static inline void tdm_load_2d(unsigned lds_addr, const void* gptr,
                                          unsigned tensor_d0, unsigned tensor_d1,
                                          unsigned tile_d0, unsigned tile_d1,
                                          unsigned long long stride0) {
    const unsigned long long ga = (unsigned long long)(uintptr_t)gptr;
    v4u g0;
    g0[0] = 1u;
    g0[1] = lds_addr;
    g0[2] = (unsigned)ga;
    g0[3] = (unsigned)(ga >> 32) | (2u << 30);
    v8i g1;
    g1[0] = (int)(3u << 16);
    g1[1] = (int)((tensor_d0 & 0xFFFFu) << 16);
    g1[2] = (int)((tensor_d0 >> 16) | ((tensor_d1 & 0xFFFFu) << 16));
    g1[3] = (int)((tensor_d1 >> 16) | (tile_d0 << 16));
    g1[4] = (int)tile_d1;
    g1[5] = (int)(unsigned)(stride0 & 0xFFFFFFFFull);
    g1[6] = (int)(unsigned)((stride0 >> 32) & 0xFFFFull);
    g1[7] = 0;
    v4i gz = {0, 0, 0, 0};
#if defined(__clang_major__) && (__clang_major__ >= 23)
    v8i z8 = {0, 0, 0, 0, 0, 0, 0, 0};
    __builtin_amdgcn_tensor_load_to_lds(g0, g1, gz, gz, z8, 0);
#else
    __builtin_amdgcn_tensor_load_to_lds(g0, g1, gz, gz, 0);
#endif
}

__global__ __launch_bounds__(256)
void zero_stats_k(float* __restrict__ stats) {
    int t = threadIdx.x;
    for (int i = t; i < 1024; i += 256) stats[i] = 0.0f;
}

__global__ __launch_bounds__(256)
void calc_alpha_k(const float* __restrict__ w, float* __restrict__ alpha) {
    __shared__ float red[256];
    const int o = blockIdx.x;
    float s = 0.0f;
    for (int i = threadIdx.x; i < 2304; i += 256)
        s += fabsf(w[(size_t)o * 2304 + i]);
    red[threadIdx.x] = s;
    __syncthreads();
    for (int st = 128; st > 0; st >>= 1) {
        if (threadIdx.x < st) red[threadIdx.x] += red[threadIdx.x + st];
        __syncthreads();
    }
    if (threadIdx.x == 0) alpha[o] = red[0] / 2304.0f;
}

__global__ __launch_bounds__(256)
void repack_weights_k(const float* __restrict__ w, char* __restrict__ wpack) {
    const int bx = blockIdx.x;
    const int ks = bx % 36;
    const int otile = bx / 36;
    const int tap = ks >> 2, cc = ks & 3;
    const int dy = tap / 3, dx = tap - dy * 3;
    const int tid = threadIdx.x;
    const int lane = tid >> 3, j = tid & 7;
    const int m = lane & 15, hh = lane >> 4;
    const int o = otile * 16 + m;
    const int kbase = ((j >> 2) << 5) + (hh << 4) + ((j & 3) << 2);
    unsigned int word = 0;
    for (int b = 0; b < 4; ++b) {
        const int c = (cc << 6) + kbase + b;
        const float v = w[(((size_t)o * CC_ + c) * 3 + dy) * 3 + dx];
        const char s = (v > 0.0f) ? 1 : ((v < 0.0f) ? -1 : 0);
        word |= ((unsigned int)(unsigned char)s) << (8 * b);
    }
    const int off = (j < 4) ? (lane * 16 + j * 4) : (512 + lane * 16 + (j - 4) * 4);
    *(int*)(wpack + ((size_t)bx << 10) + off) = word;
}

__global__ __launch_bounds__(256)
void binarize_input_k(const float* __restrict__ x, char* __restrict__ pad) {
    const size_t idx = (size_t)blockIdx.x * 256 + threadIdx.x;
    const int cg = (int)(idx & 63);
    const size_t pos = idx >> 6;
    const int wp = (int)(pos % WP);
    const size_t t = pos / WP;
    const int hp = (int)(t % HP);
    const int n = (int)(t / HP);
    unsigned int word = 0;
    if (hp >= 1 && hp <= HH_ && wp >= 1 && wp <= WW_) {
        const int h = hp - 1, w = wp - 1;
        for (int j = 0; j < 4; ++j) {
            const int c = cg * 4 + j;
            const float v = x[(((size_t)n * CC_ + c) * HH_ + h) * WW_ + w];
            const char s = (v > 0.0f) ? 1 : ((v < 0.0f) ? -1 : 0);
            word |= ((unsigned int)(unsigned char)s) << (8 * j);
        }
    }
    *(int*)(pad + pos * 256 + cg * 4) = (int)word;
}

__global__ __launch_bounds__(256)
void conv_wmma_k(const char*  __restrict__ pad,
                 const char*  __restrict__ wpack,
                 const float* __restrict__ alpha,
                 float*       __restrict__ y,
                 float*       __restrict__ stats) {
    extern __shared__ __align__(16) char smem[];
    char*  sW = smem;
    char*  sX = smem + LDSW_BYTES;
    float* sS = (float*)(smem + LDSW_BYTES + LDSX_BYTES);

    const int tid = threadIdx.x;
    const int row = blockIdx.x;
    const int og  = blockIdx.y;
    const int n = row / HH_, h = row % HH_;

    if (tid < 64) sS[tid] = 0.0f;

    const int wv = tid >> 5;
    if (wv == 0) {
        tdm_load_2d((unsigned)(uintptr_t)sW,
                    wpack + (size_t)og * LDSW_BYTES,
                     LDSW_BYTES / 8,  1,
                     LDSW_BYTES / 8,    1,
                     LDSW_BYTES / 8);
        __builtin_amdgcn_s_wait_tensorcnt(0);
    } else if (wv == 1) {
        tdm_load_2d((unsigned)(uintptr_t)sX,
                    pad + ((((size_t)n * HP + h) * WP) << 8),
                     (WP * 256) / 8,  3,
                     LDSX_ROW / 8,      3,
                     (WP * 256) / 8);
        __builtin_amdgcn_s_wait_tensorcnt(0);
    }
    __syncthreads();

    const int lane  = tid & 31;
    const int ptile = wv & 3;
    const int ot    = wv >> 2;
    const int kh    = lane >> 4;
    const int ncol  = lane & 15;

    v8i acc = {0, 0, 0, 0, 0, 0, 0, 0};
#pragma unroll
    for (int ks = 0; ks < KSTEPS; ++ks) {
        const int tap = ks >> 2, cc = ks & 3;
        const int dy = tap / 3, dx = tap - dy * 3;
        const char* aw = sW + ((ot * 36 + ks) << 10) + (lane << 4);
        const v4i a_lo = *(const v4i*)(aw);
        const v4i a_hi = *(const v4i*)(aw + 512);
        const char* bx = sX + dy * LDSX_ROW
                       + ((ptile * 16 + ncol + dx) << 8) + (cc << 6) + (kh << 4);
        const v4i b_lo = *(const v4i*)(bx);
        const v4i b_hi = *(const v4i*)(bx + 32);
        const v8i A = __builtin_shufflevector(a_lo, a_hi, 0, 1, 2, 3, 4, 5, 6, 7);
        const v8i B = __builtin_shufflevector(b_lo, b_hi, 0, 1, 2, 3, 4, 5, 6, 7);
        acc = __builtin_amdgcn_wmma_i32_16x16x64_iu8(
             true, A,  true, B, acc,
             false,  false);
    }

    const int w = ptile * 16 + ncol;
    if (w < WW_) {
        const int olocal = ot * 16 + kh * 8;
        const int obase  = og * 32 + olocal;
        v4f f0, f1;
#pragma unroll
        for (int r = 0; r < 4; ++r) f0[r] = alpha[obase + r]     * (float)acc[r];
#pragma unroll
        for (int r = 0; r < 4; ++r) f1[r] = alpha[obase + 4 + r] * (float)acc[4 + r];
#pragma unroll
        for (int r = 0; r < 4; ++r) {
            atomicAdd(&sS[olocal + r],          f0[r]);
            atomicAdd(&sS[32 + olocal + r],     f0[r] * f0[r]);
            atomicAdd(&sS[olocal + 4 + r],      f1[r]);
            atomicAdd(&sS[32 + olocal + 4 + r], f1[r] * f1[r]);
        }
        const size_t ybase = (((size_t)row) * WW_ + w) * 256 + obase;
        *(v4f*)(y + ybase)     = f0;
        *(v4f*)(y + ybase + 4) = f1;
    }
    __syncthreads();
    if (tid < 64) {
        const int o = og * 32 + (tid & 31);
        float* dst = stats + ((tid < 32) ? 0 : 256) + o;
        atomicAdd(dst, sS[tid]);
    }
}

__global__ __launch_bounds__(256)
void bn_finalize_k(const float* __restrict__ stats,
                   const float* __restrict__ gamma,
                   const float* __restrict__ beta,
                   float* __restrict__ ss) {
    const int c = threadIdx.x;
    const float inv = 1.0f / (float)NPIX;
    const float mean = stats[c] * inv;
    const float var  = stats[256 + c] * inv - mean * mean;
    const float sc   = gamma[c] * rsqrtf(var + 1e-5f);
    ss[c]       = sc;
    ss[256 + c] = beta[c] - mean * sc;
}

__global__ __launch_bounds__(256)
void bn_binarize_k(const float* __restrict__ y, const float* __restrict__ ss,
                   char* __restrict__ pad) {
    const size_t idx = (size_t)blockIdx.x * 256 + threadIdx.x;
    const int cg = (int)(idx & 63);
    const size_t pos = idx >> 6;
    const int wp = (int)(pos % WP);
    const size_t t = pos / WP;
    const int hp = (int)(t % HP);
    const int n = (int)(t / HP);
    unsigned int word = 0;
    if (hp >= 1 && hp <= HH_ && wp >= 1 && wp <= WW_) {
        const int h = hp - 1, w = wp - 1;
        const size_t base = (((size_t)n * HH_ + h) * WW_ + w) * 256 + cg * 4;
        const v4f v = *(const v4f*)(y + base);
        for (int j = 0; j < 4; ++j) {
            const int c = cg * 4 + j;
            const float z = v[j] * ss[c] + ss[256 + c];
            const char s = (z > 0.0f) ? 1 : ((z < 0.0f) ? -1 : 0);
            word |= ((unsigned int)(unsigned char)s) << (8 * j);
        }
    }
    *(int*)(pad + pos * 256 + cg * 4) = (int)word;
}

__global__ __launch_bounds__(256)
void final_add_k(const float* __restrict__ x, const float* __restrict__ y,
                 const float* __restrict__ ss, float* __restrict__ out) {
    const size_t idx = (size_t)blockIdx.x * 256 + threadIdx.x;
    const int w4 = (int)(idx % 14);
    size_t t = idx / 14;
    const int h = (int)(t % HH_); t /= HH_;
    const int c = (int)(t % CC_);
    const int n = (int)(t / CC_);
    const size_t xb = (((size_t)n * CC_ + c) * HH_ + h) * WW_ + w4 * 4;
    const v4f xv = *(const v4f*)(x + xb);
    const float sc = ss[c], sh = ss[256 + c];
    v4f o;
#pragma unroll
    for (int i = 0; i < 4; ++i) {
        const float yy = y[(((size_t)n * HH_ + h) * WW_ + w4 * 4 + i) * 256 + c];
        o[i] = yy * sc + sh + xv[i];
    }
    *(v4f*)(out + xb) = o;
}

extern "C" void kernel_launch(void* const* d_in, const int* in_sizes, int n_in,
                              void* d_out, int out_size, void* d_ws, size_t ws_size,
                              hipStream_t stream) {
    (void)in_sizes; (void)n_in; (void)out_size; (void)ws_size;
    const float* x  = (const float*)d_in[0];
    const float* w1 = (const float*)d_in[1];
    const float* g1 = (const float*)d_in[2];
    const float* b1 = (const float*)d_in[3];
    const float* w2 = (const float*)d_in[4];
    const float* g2 = (const float*)d_in[5];
    const float* b2 = (const float*)d_in[6];
    float* out = (float*)d_out;

    char* ws = (char*)d_ws;
    size_t off = 0;
    char*  pad    = ws + off; off += (size_t)NN * HP * WP * 256;
    float* ybuf   = (float*)(ws + off); off += (size_t)NPIX * 256 * 4;
    char*  wpack1 = ws + off; off += 16 * 36 * 1024;
    char*  wpack2 = ws + off; off += 16 * 36 * 1024;
    float* alpha1 = (float*)(ws + off); off += 256 * 4;
    float* alpha2 = (float*)(ws + off); off += 256 * 4;
    float* stats1 = (float*)(ws + off); off += 512 * 4;
    float* stats2 = (float*)(ws + off); off += 512 * 4;
    float* ss1    = (float*)(ws + off); off += 512 * 4;
    float* ss2    = (float*)(ws + off); off += 512 * 4;

    const int pad_blocks = (int)(((size_t)NN * HP * WP * 64) / 256);
    const int add_blocks = (int)(((size_t)NN * CC_ * HH_ * (WW_ / 4)) / 256);
    const dim3 conv_grid(NN * HH_, 8);
    const size_t conv_smem = LDSW_BYTES + LDSX_BYTES + 64 * 4;

    zero_stats_k<<<1, 256, 0, stream>>>(stats1);

    calc_alpha_k<<<256, 256, 0, stream>>>(w1, alpha1);
    calc_alpha_k<<<256, 256, 0, stream>>>(w2, alpha2);
    repack_weights_k<<<576, 256, 0, stream>>>(w1, wpack1);
    repack_weights_k<<<576, 256, 0, stream>>>(w2, wpack2);

    binarize_input_k<<<pad_blocks, 256, 0, stream>>>(x, pad);
    conv_wmma_k<<<conv_grid, 256, conv_smem, stream>>>(pad, wpack1, alpha1, ybuf, stats1);
    bn_finalize_k<<<1, 256, 0, stream>>>(stats1, g1, b1, ss1);

    bn_binarize_k<<<pad_blocks, 256, 0, stream>>>(ybuf, ss1, pad);
    conv_wmma_k<<<conv_grid, 256, conv_smem, stream>>>(pad, wpack2, alpha2, ybuf, stats2);
    bn_finalize_k<<<1, 256, 0, stream>>>(stats2, g2, b2, ss2);

    final_add_k<<<add_blocks, 256, 0, stream>>>(x, ybuf, ss2, out);
}
